// SpectralConv1d_74972949119501
// MI455X (gfx1250) — matrix operand order repaired and checked by reading, not compiled, not run on hardware
//
#include <hip/hip_runtime.h>

// ---------------------------------------------------------------------------
// SpectralConv1d as three GEMMs on CDNA5 WMMA (f16 in, f32 accumulate).
//   Stage A: X[8192x128]  = q-rows[8192x2048] * F[2048x128]   (partial DFT)
//   Stage B: Z[8192x128]  = complex mix per (h,m): (16x64)x(64x64)
//   Stage C: y[8192x2048] = Z[8192x128] * G[128x2048]         (trunc. irFFT)
// Stage A: double-buffered LDS A-tile (32x32), 1 barrier / K-step.
// Stage C: A-tile (32x128 f32) fetched by the Tensor Data Mover into LDS.
// ---------------------------------------------------------------------------

typedef __attribute__((ext_vector_type(16))) _Float16 v16h;
typedef __attribute__((ext_vector_type(8)))  _Float16 v8h;
typedef __attribute__((ext_vector_type(8)))  float    v8f;
typedef __attribute__((ext_vector_type(4)))  unsigned int v4u;
typedef __attribute__((ext_vector_type(8)))  int      v8i;
typedef __attribute__((ext_vector_type(4)))  int      v4i;

struct alignas(32) HPair { v8h lo, hi; };

static __device__ __forceinline__ v16h joinh(v8h lo, v8h hi) {
  HPair p{lo, hi};
  return __builtin_bit_cast(v16h, p);
}

// Pack a 16-element f16 A/B fragment from two runs of 8 f32 (LDS resident).
static __device__ __forceinline__ v16h cvt16(const float* p0, const float* p1) {
  v16h r;
#pragma unroll
  for (int e = 0; e < 8; ++e) {
    r[e]     = (_Float16)p0[e];
    r[e + 8] = (_Float16)p1[e];
  }
  return r;
}

static __device__ __forceinline__ v8f wmma_f16(v16h a, v16h b, v8f c) {
  // D = A(16x32 f16) x B(32x16 f16) + C(16x16 f32)
  return __builtin_amdgcn_wmma_f32_16x16x32_f16(
      /*neg_a=*/false, a, /*neg_b=*/false, b,
      /*c_mod=*/(short)0, c, /*reuse_a=*/false, /*reuse_b=*/false);
}

// Problem constants
#define NB   16
#define NL   2048
#define NH   8
#define NE   64
#define NM   64
#define NR   (NB*NH*NE)      // 8192 rows = (b,h,chan)
#define TWO_PI_OVER_L 0.003067961575771282f   // 2*pi/2048

// ---------------------------------------------------------------------------
// Twiddle tables (f16, K-contiguous per output column for B-fragment loads)
//   T1t [128][2048]: row n<64 -> cos(2pi n l / L); n>=64 -> -sin(2pi (n-64) l / L)
//   T2  [2048][128]: col k<64 -> a_k cos(2pi k l / L); k>=64 -> -a sin(...)
//                    a_0 = 1/L, a_m = 2/L  (truncated irfft weights)
// ---------------------------------------------------------------------------
__global__ void __launch_bounds__(256)
spc_build_tables(_Float16* __restrict__ T1t, _Float16* __restrict__ T2) {
  int idx = blockIdx.x * 256 + threadIdx.x;          // 0 .. 2048*128-1
  {
    int l = idx & 2047, n = idx >> 11;
    int ph = ((n & 63) * l) & 2047;                  // periodic reduction
    float w = TWO_PI_OVER_L * (float)ph;
    float s, c; __sincosf(w, &s, &c);
    T1t[idx] = (_Float16)((n < 64) ? c : -s);
  }
  {
    int k = idx & 127, l = idx >> 7;
    int ph = ((k & 63) * l) & 2047;
    float w = TWO_PI_OVER_L * (float)ph;
    float s, c; __sincosf(w, &s, &c);
    float a = (((k & 63) == 0) ? 1.0f : 2.0f) * (1.0f / 2048.0f);
    T2[idx] = (_Float16)(a * ((k < 64) ? c : -s));
  }
}

// ---------------------------------------------------------------------------
// Stage A: partial DFT.  Block = 256 thr (8 waves) owns 32 rows x 128 cols.
// Double-buffered 32x32 f16 LDS A-tile; 1 barrier per K-step; each wave does
// 2 WMMAs (row halves) per 32B B-fragment load.
// q layout (B,L,H,E): row r=(b,h,i) -> base (r>>9)*1048576 + (r&511), +l*512.
// ---------------------------------------------------------------------------
__global__ void __launch_bounds__(256)
spc_stageA(const float* __restrict__ q, const _Float16* __restrict__ T1t,
           float* __restrict__ Xg) {
  __shared__ alignas(16) _Float16 As[2][32][32];
  const int t = threadIdx.x;
  const int lane = t & 31, wave = t >> 5;
  const int rowbase = blockIdx.x * 32;

  // cooperative-load mapping: 32 rows x 32 K, 4 elements per thread
  const int rloc = t & 31, kpart = t >> 5;           // kpart in [0,8)
  const int r = rowbase + rloc;
  const float* qrow = q + (size_t)(r >> 9) * (NL * 512) + (r & 511);

  // per-wave B column + per-lane fragment bases
  const int col  = wave * 16 + (lane & 15);
  const _Float16* Bcol = T1t + (size_t)col * NL + ((lane < 16) ? 0 : 8);
  const int khalf = (lane < 16) ? 0 : 8;

  // prologue: K-step 0 into buffer 0
#pragma unroll
  for (int j = 0; j < 4; ++j)
    As[0][rloc][kpart + 8 * j] = (_Float16)qrow[(size_t)(kpart + 8 * j) * 512];

  v8f acc0 = {}, acc1 = {};
  for (int ks = 0; ks < 64; ++ks) {
    const int buf = ks & 1;
    const int l0 = ks * 32;
    __syncthreads();                         // buf[ks&1] now ready
    if (ks < 63) {                           // overlap: fetch next K-step
      const int l1 = l0 + 32;
#pragma unroll
      for (int j = 0; j < 4; ++j)
        As[buf ^ 1][rloc][kpart + 8 * j] =
            (_Float16)qrow[(size_t)(l1 + kpart + 8 * j) * 512];
      __builtin_prefetch(qrow + (size_t)(l1 + 32 + kpart) * 512, 0, 3);
    }
    v8h blo = *(const v8h*)(Bcol + l0);
    v8h bhi = *(const v8h*)(Bcol + l0 + 16);
    v16h bf = joinh(blo, bhi);
    const _Float16* a0 = &As[buf][lane & 15][khalf];
    const _Float16* a1 = &As[buf][16 + (lane & 15)][khalf];
    acc0 = wmma_f16(joinh(*(const v8h*)a0, *(const v8h*)(a0 + 16)), bf, acc0);
    acc1 = wmma_f16(joinh(*(const v8h*)a1, *(const v8h*)(a1 + 16)), bf, acc1);
  }

  const int mofs = (lane < 16) ? 0 : 8;
  float* out0 = Xg + (size_t)(rowbase + mofs) * 128 + col;
  float* out1 = Xg + (size_t)(rowbase + 16 + mofs) * 128 + col;
#pragma unroll
  for (int j = 0; j < 8; ++j) {
    out0[(size_t)j * 128] = acc0[j];
    out1[(size_t)j * 128] = acc1[j];
  }
}

// ---------------------------------------------------------------------------
// Stage B: per (h,m) complex mix. Block = 128 thr (4 waves), one (h,m) each.
// ---------------------------------------------------------------------------
__global__ void __launch_bounds__(128)
spc_stageB(const float* __restrict__ Wr, const float* __restrict__ Wi,
           const float* __restrict__ Xg, float* __restrict__ Z) {
  const int t = threadIdx.x, lane = t & 31, wave = t >> 5;
  const int h = blockIdx.x >> 6, m = blockIdx.x & 63;
  const int brow = lane & 15;                        // batch = A-matrix row
  const int o = wave * 16 + (lane & 15);             // output channel = B col

  const float* Xbase = Xg + ((size_t)brow * 512 + h * 64) * 128 + m;   // + i*128
  const float* WrB = Wr + ((size_t)h * 4096 + o) * 64 + m;             // + i*4096
  const float* WiB = Wi + ((size_t)h * 4096 + o) * 64 + m;

  v8f pRR = {}, pII = {}, pRI = {}, pIR = {};
#pragma unroll
  for (int ks = 0; ks < 2; ++ks) {
    const int kA = ks * 32 + ((lane < 16) ? 0 : 8);
    const int kB = ks * 32 + ((lane < 16) ? 0 : 8);
    v16h ar, ai, br, bi;
#pragma unroll
    for (int e = 0; e < 8; ++e) {
      const int i1 = kA + e, i2 = kA + 16 + e;
      ar[e]     = (_Float16)Xbase[(size_t)i1 * 128];
      ar[e + 8] = (_Float16)Xbase[(size_t)i2 * 128];
      ai[e]     = (_Float16)Xbase[(size_t)i1 * 128 + 64];
      ai[e + 8] = (_Float16)Xbase[(size_t)i2 * 128 + 64];
    }
#pragma unroll
    for (int e = 0; e < 16; ++e) {
      const int i = kB + e + ((e >= 8) ? 8 : 0);
      br[e] = (_Float16)(4096.0f * WrB[(size_t)i * 4096]);  // into f16 normal range
      bi[e] = (_Float16)(4096.0f * WiB[(size_t)i * 4096]);
    }
    pRR = wmma_f16(ar, br, pRR);
    pII = wmma_f16(ai, bi, pII);
    pRI = wmma_f16(ar, bi, pRI);
    pIR = wmma_f16(ai, br, pIR);
  }

  const int mofs = (lane < 16) ? 0 : 8;
#pragma unroll
  for (int j = 0; j < 8; ++j) {
    const int b = j + mofs;
    const size_t zr = ((size_t)b * 512 + (size_t)h * 64 + o) * 128;
    Z[zr + m]      = (pRR[j] - pII[j]) * (1.0f / 4096.0f);   // Re
    Z[zr + m + 64] = (pRI[j] + pIR[j]) * (1.0f / 4096.0f);   // Im
  }
}

// ---------------------------------------------------------------------------
// Stage C: truncated irFFT. Block = 256 thr (8 waves) owns 32 rows x 128 cols
// of y. The 32x128 f32 A-tile is DMA'd Global->LDS by the Tensor Data Mover
// (one tensor_load_to_lds issued by wave 0, published via TENSORcnt+barrier);
// f32->f16 conversion happens at fragment-pack time and co-executes with WMMA.
// ---------------------------------------------------------------------------
__global__ void __launch_bounds__(256)
spc_stageC(const float* __restrict__ Z, const _Float16* __restrict__ T2,
           float* __restrict__ Y) {
  __shared__ alignas(16) float AsC[32][128];
  const int t = threadIdx.x, lane = t & 31, wave = t >> 5;
  const int rowbase = (blockIdx.x >> 4) * 32;
  const int coltile = blockIdx.x & 15;

#if __has_builtin(__builtin_amdgcn_tensor_load_to_lds)
  if (wave == 0) {
    const unsigned lds_addr = (unsigned)(uintptr_t)(&AsC[0][0]);
    const unsigned long long ga =
        (unsigned long long)(uintptr_t)(Z + (size_t)rowbase * 128);
    // D# group0: count=1 | lds_addr | global_addr[56:0] | type=2
    v4u g0 = { 1u,
               lds_addr,
               (unsigned)(ga & 0xffffffffu),
               (unsigned)((ga >> 32) & 0x1ffffffu) | (2u << 30) };
    // D# group1: data_size=4B; tensor 128 x 8192; tile 128 x 32; stride 128
    v8i g1 = { (int)0x20000u,        // [17:16] data_size = 2 (4 bytes)
               (int)(128u << 16),    // tensor_dim0 = 128       (bits 79:48)
               (int)(8192u << 16),   // tensor_dim1 = 8192      (bits 111:80)
               (int)(128u << 16),    // tile_dim0 = 128         (bits 127:112)
               32,                   // tile_dim1 = 32          (bits 143:128)
               128,                  // tensor_dim0_stride = 128 (bits 207:160)
               0, 0 };
    v4i g2 = { 0, 0, 0, 0 }, g3 = { 0, 0, 0, 0 };
    v8i g4 = { 0, 0, 0, 0, 0, 0, 0, 0 };   // unused trailing descriptor group
    __builtin_amdgcn_tensor_load_to_lds(g0, g1, g2, g3, g4, 0);
    __builtin_amdgcn_s_wait_tensorcnt(0);
  }
#else
  {  // fallback: cooperative load of the 32x128 f32 A-tile
    const int e = t * 16;
    const int rr = e >> 7, kk = e & 127;
    const float* zp = Z + (size_t)(rowbase + rr) * 128 + kk;
#pragma unroll
    for (int j = 0; j < 16; ++j) AsC[rr][kk + j] = zp[j];
  }
#endif
  __syncthreads();

  const int col = coltile * 128 + wave * 16 + (lane & 15);   // output sample l
  const _Float16* Bcol = T2 + (size_t)col * 128 + ((lane < 16) ? 0 : 8);
  const float* arow0 = &AsC[lane & 15][0];
  const float* arow1 = &AsC[16 + (lane & 15)][0];
  const int khalf = (lane < 16) ? 0 : 8;

  v8f acc0 = {}, acc1 = {};
#pragma unroll
  for (int ks = 0; ks < 4; ++ks) {
    const int l0 = ks * 32;
    v16h bf = joinh(*(const v8h*)(Bcol + l0), *(const v8h*)(Bcol + l0 + 16));
    acc0 = wmma_f16(cvt16(arow0 + l0 + khalf, arow0 + l0 + khalf + 16), bf, acc0);
    acc1 = wmma_f16(cvt16(arow1 + l0 + khalf, arow1 + l0 + khalf + 16), bf, acc1);
  }

  const int mofs = (lane < 16) ? 0 : 8;
  float* y0 = Y + (size_t)(rowbase + mofs) * NL + col;
  float* y1 = Y + (size_t)(rowbase + 16 + mofs) * NL + col;
#pragma unroll
  for (int j = 0; j < 8; ++j) {
    y0[(size_t)j * NL] = acc0[j];
    y1[(size_t)j * NL] = acc1[j];
  }
}

// ---------------------------------------------------------------------------
extern "C" void kernel_launch(void* const* d_in, const int* in_sizes, int n_in,
                              void* d_out, int out_size, void* d_ws, size_t ws_size,
                              hipStream_t stream) {
  const float* q  = (const float*)d_in[0];   // (16,2048,8,64)
  const float* Wr = (const float*)d_in[1];   // (8,64,64,64)
  const float* Wi = (const float*)d_in[2];   // (8,64,64,64)
  float* y = (float*)d_out;                  // (16,8,64,2048)

  char* ws = (char*)d_ws;
  _Float16* T1t = (_Float16*)(ws);                          // 128*2048 f16 = 512KB
  _Float16* T2  = (_Float16*)(ws + (512u << 10));           // 2048*128 f16 = 512KB
  float*    Xg  = (float*)(ws + (1024u << 10));             // 8192*128 f32 = 4MB
  float*    Z   = (float*)(ws + (1024u << 10) + (4u << 20));// 8192*128 f32 = 4MB

  spc_build_tables<<<dim3(1024), dim3(256), 0, stream>>>(T1t, T2);
  spc_stageA<<<dim3(NR / 32), dim3(256), 0, stream>>>(q, T1t, Xg);
  spc_stageB<<<dim3(NH * NM), dim3(128), 0, stream>>>(Wr, Wi, Xg, Z);
  spc_stageC<<<dim3((NR / 32) * 16), dim3(256), 0, stream>>>(Z, T2, y);
}
